// SLENet_56255481643648
// MI455X (gfx1250) — matrix operand order repaired and checked by reading, not compiled, not run on hardware
//
#include <hip/hip_runtime.h>
#include <hip/hip_bf16.h>

typedef __attribute__((ext_vector_type(16))) _Float16 v16h;
typedef __attribute__((ext_vector_type(8)))  float    v8f;
typedef __attribute__((ext_vector_type(4)))  unsigned int v4u;
typedef __attribute__((ext_vector_type(4)))  int      v4i;
typedef __attribute__((ext_vector_type(8)))  int      v8i;
typedef unsigned int u32;

// ---------------------------------------------------------------------------
// WMMA tile helpers (CDNA5 wave32, V_WMMA_F32_16X16X32_F16)
//
// A (16x32 f16): lane = 16*khalf + row ; dword j holds K pair
//     k = khalf*16 + j*2                  (+0/+1 in low/high half)
// B (32x16 f16): lane = 16*khalf + col ; dword j holds K pair
//     k = khalf*16 + j*2                  (+0/+1)
// C/D (16x16 f32): lane = 16*mhalf + col ; VGPR v -> row = mhalf*8 + v
// ---------------------------------------------------------------------------

union AFrag { v16h v; _Float16 h[16]; };
union BFrag { v16h v; uint4 q[2]; };

__device__ __forceinline__ void wave_gemm(const float* __restrict__ src, int ldi, int nkb,
                                          const u32* __restrict__ bpack, int nnb,
                                          float* __restrict__ dst, int ldo,
                                          int wrow, int lane)
{
    v16h a[6];
    const int row   = wrow + (lane & 15);
    const int khalf = lane >> 4;
    for (int kb = 0; kb < nkb; ++kb) {
        AFrag af;
#pragma unroll
        for (int j = 0; j < 8; ++j) {
            const int k = kb * 32 + khalf * 16 + j * 2;
            af.h[2 * j]     = (_Float16)src[row * ldi + k];
            af.h[2 * j + 1] = (_Float16)src[row * ldi + k + 1];
        }
        a[kb] = af.v;
    }
    for (int nb = 0; nb < nnb; ++nb) {
        v8f acc = {0.f, 0.f, 0.f, 0.f, 0.f, 0.f, 0.f, 0.f};
        for (int kb = 0; kb < nkb; ++kb) {
            BFrag bf;
            const u32* p = bpack + ((size_t)(nb * nkb + kb) * 32 + lane) * 8;
            bf.q[0] = *(const uint4*)p;
            bf.q[1] = *(const uint4*)(p + 4);
            acc = __builtin_amdgcn_wmma_f32_16x16x32_f16(
                false, a[kb], false, bf.v, (short)0, acc, false, false);
        }
        const int drow = wrow + (lane >> 4) * 8;
        const int dcol = nb * 16 + (lane & 15);
#pragma unroll
        for (int v = 0; v < 8; ++v) dst[(drow + v) * ldo + dcol] = acc[v];
    }
}

// ---------------------------------------------------------------------------
// Tensor Data Mover: load a [rows x 96] f32 tile (row-major, contiguous rows)
// from global into LDS (contiguous) with a single TDM descriptor.
// D# bitfields per CDNA5 ISA 8.3/8.4. Issue from one wave; EXEC is ignored.
// ---------------------------------------------------------------------------
__device__ __forceinline__ void tdm_load_tile_f32(const float* gsrc, u32 lds_byte_off,
                                                  int rows, int cols, unsigned totRows)
{
    const unsigned long long ga = (unsigned long long)(uintptr_t)gsrc;
    v4u g0;
    g0[0] = 1u;                                        // count=1 valid user D#
    g0[1] = lds_byte_off;                              // lds_addr
    g0[2] = (u32)(ga & 0xffffffffull);                 // global_addr[31:0]
    g0[3] = (u32)((ga >> 32) & 0x1ffffffull)           // global_addr[56:32]
          | (2u << 30);                                // type=2 (image)
    v8i g1;
    g1[0] = 0x00020000;                                // data_size=2 (4 bytes)
    g1[1] = (int)((u32)cols << 16);                    // tensor_dim0[15:0]
    g1[2] = (int)((totRows & 0xffffu) << 16);          // dim0[31:16]=0 | dim1[15:0]
    g1[3] = (int)(((totRows >> 16) & 0xffffu)          // tensor_dim1[31:16]
          | ((u32)cols << 16));                        // tile_dim0
    g1[4] = rows;                                      // tile_dim1 | tile_dim2=0
    g1[5] = cols;                                      // tensor_dim0_stride[31:0]
    g1[6] = 0;                                         // stride hi | dim1_stride lo
    g1[7] = 0;                                         // dim1_stride hi
    v4i g2 = {0, 0, 0, 0};
    v4i g3 = {0, 0, 0, 0};
#if __has_include(<hip/amd_detail/amd_gfx1250_TDM.h>)
    v8i g4 = {0, 0, 0, 0, 0, 0, 0, 0};
    __builtin_amdgcn_tensor_load_to_lds(g0, g1, g2, g3, g4, 0);
#else
    __builtin_amdgcn_tensor_load_to_lds(g0, g1, g2, g3, 0);
#endif
}

// Per-row LayerNorm over C columns starting at col0 of an LDS buffer.
__device__ __forceinline__ void ln_rows(float* buf, int ld, int col0, int C,
                                        const float* __restrict__ w,
                                        const float* __restrict__ b,
                                        int tid, int nthr, int rows)
{
    for (int r = tid; r < rows; r += nthr) {
        float* p = buf + r * ld + col0;
        float mu = 0.f;
        for (int c = 0; c < C; ++c) mu += p[c];
        mu /= (float)C;
        float var = 0.f;
        for (int c = 0; c < C; ++c) { float d = p[c] - mu; var += d * d; }
        var /= (float)C;
        const float inv = rsqrtf(var + 1e-5f);
        for (int c = 0; c < C; ++c) p[c] = (p[c] - mu) * inv * w[c] + b[c];
    }
}

// ---------------------------------------------------------------------------
// Kernel: zero scratch
// ---------------------------------------------------------------------------
__global__ void zero_kernel(float* p, int n)
{
    int i = blockIdx.x * blockDim.x + threadIdx.x;
    if (i < n) p[i] = 0.f;
}

// ---------------------------------------------------------------------------
// Kernel: pack a torch-layout weight W[N(out)][K(in)] (f32) into f16 WMMA
// B-fragments for computing  x @ W^T   (B[k][n] = W[n][k]).
// ---------------------------------------------------------------------------
__global__ void pack_weight_kernel(const float* __restrict__ W, u32* __restrict__ dst,
                                   int Kt, int Nt)
{
    const int nkb = Kt / 32, nnb = Nt / 16;
    const int total = nnb * nkb * 32 * 8;
    for (int idx = blockIdx.x * blockDim.x + threadIdx.x; idx < total;
         idx += gridDim.x * blockDim.x) {
        const int j    = idx & 7;
        const int lane = (idx >> 3) & 31;
        const int f    = idx >> 8;
        const int kb = f % nkb, nb = f / nkb;
        const int n = nb * 16 + (lane & 15);
        const int k = kb * 32 + (lane >> 4) * 16 + j * 2;
        union { _Float16 h[2]; u32 u; } pk;
        pk.h[0] = (_Float16)W[(size_t)n * Kt + k];
        pk.h[1] = (_Float16)W[(size_t)n * Kt + k + 1];
        dst[((size_t)f * 32 + lane) * 8 + j] = pk.u;
    }
}

// ---------------------------------------------------------------------------
// Kernel: pack KV[8][12][12] (f32) into block-diagonal 96x96 B-fragments so
// msg_pre = Q @ KVblk is a normal WMMA GEMM.
// ---------------------------------------------------------------------------
__global__ void pack_kv_kernel(const float* __restrict__ KV, u32* __restrict__ dst)
{
    const int nkb = 3;
    const int total = 6 * 3 * 32 * 8;
    for (int idx = blockIdx.x * blockDim.x + threadIdx.x; idx < total;
         idx += gridDim.x * blockDim.x) {
        const int j    = idx & 7;
        const int lane = (idx >> 3) & 31;
        const int f    = idx >> 8;
        const int kb = f % nkb, nb = f / nkb;
        const int n = nb * 16 + (lane & 15);
        const int k = kb * 32 + (lane >> 4) * 16 + j * 2;
        union { _Float16 h[2]; u32 u; } pk;
#pragma unroll
        for (int t = 0; t < 2; ++t) {
            const int kk = k + t;
            const int h = kk / 12;
            float v = 0.f;
            if (n / 12 == h) v = KV[h * 144 + (kk % 12) * 12 + (n % 12)];
            pk.h[t] = (_Float16)v;
        }
        dst[((size_t)f * 32 + lane) * 8 + j] = pk.u;
    }
}

// ---------------------------------------------------------------------------
// Kernel: linear-attention statistics.
// KV[h][d][e] = sum_s (elu(k)+1)[s,h,d] * v[s,h,e] / S ;  Ksum[h][d] = sum_s K
// One block = 64 rows, 4 waves; tile in via TDM, projections via WMMA,
// reduction via VALU + global f32 atomics.
// ---------------------------------------------------------------------------
__global__ void __launch_bounds__(128)
attn_stats_kernel(const float* __restrict__ warp,
                  const u32* __restrict__ wkpack, const u32* __restrict__ wvpack,
                  float* __restrict__ KVout, float* __restrict__ Ksumout,
                  float invS, unsigned Srows)
{
    __shared__ float sW[64 * 96];
    __shared__ float sK[64 * 96];
    __shared__ float sV[64 * 96];
    const int tid = threadIdx.x, lane = tid & 31, wrow = (tid >> 5) * 16;
    const long tile0 = (long)blockIdx.x * 64;

    if (tid < 32)
        tdm_load_tile_f32(warp + tile0 * 96, (u32)(uintptr_t)&sW[0], 64, 96, Srows);
    if (blockIdx.x + 1 < gridDim.x)
        __builtin_prefetch((const char*)(warp + (tile0 + 64) * 96) + tid * 64, 0, 1);
    __builtin_amdgcn_s_wait_tensorcnt(0);
    __syncthreads();

    wave_gemm(sW, 96, 3, wkpack, 6, sK, 96, wrow, lane);
    wave_gemm(sW, 96, 3, wvpack, 6, sV, 96, wrow, lane);
    __syncthreads();

    for (int i = tid; i < 64 * 96; i += 128) {
        float x = sK[i];
        sK[i] = x > 0.f ? x + 1.f : __expf(x);   // elu(x)+1
    }
    __syncthreads();

    for (int e = tid; e < 1152; e += 128) {
        const int h = e / 144, rem = e % 144, d = rem / 12, c = rem % 12;
        float acc = 0.f;
        for (int r = 0; r < 64; ++r)
            acc += sK[r * 96 + h * 12 + d] * sV[r * 96 + h * 12 + c];
        atomicAdd(&KVout[e], acc * invS);
    }
    if (tid < 96) {
        float acc = 0.f;
        for (int r = 0; r < 64; ++r) acc += sK[r * 96 + tid];
        atomicAdd(&Ksumout[tid], acc);
    }
}

// ---------------------------------------------------------------------------
// Main fused kernel: 64 rows per block, tile in via TDM, all GEMMs via WMMA.
// ---------------------------------------------------------------------------
struct P2 {
    const float *mr;
    const float *ln1w, *ln1b, *ln2w, *ln2b, *fc1b, *fc2b;
    const float *saln1w, *saln1b, *saln2w, *saln2b;
    const float *ksum;
    const u32 *wqkv, *wqkv2, *wq, *merge, *kvblk, *w1, *w2, *fc1, *fc2;
    float *out;
};

__global__ void __launch_bounds__(128) fused_main(P2 p, float Sval, unsigned Lrows)
{
    __shared__ float sS[64 * 96];    // shortcut
    __shared__ float sY[64 * 96];    // y
    __shared__ float sU[64 * 192];   // dual-purpose 96+96 staging
    __shared__ float sV[64 * 192];   // x0 landing zone, later 192-wide temps
    __shared__ float sZ[64 * 8];

    const int tid = threadIdx.x, lane = tid & 31, wrow = (tid >> 5) * 16;
    const long tile0 = (long)blockIdx.x * 64;

    // ---- TDM: mr tile -> sV[0:64*96] (contiguous) ----
    if (tid < 32)
        tdm_load_tile_f32(p.mr + tile0 * 96, (u32)(uintptr_t)&sV[0], 64, 96, Lrows);
    if (blockIdx.x + 1 < gridDim.x)
        __builtin_prefetch((const char*)(p.mr + (tile0 + 64) * 96) + tid * 64, 0, 1);
    __builtin_amdgcn_s_wait_tensorcnt(0);
    __syncthreads();

    // ---- LN1 -> x0 (in place in sV) ----
    ln_rows(sV, 96, 0, 96, p.ln1w, p.ln1b, tid, 128, 64);
    __syncthreads();

    // ---- shortcut = x0 @ Wqkv^T ; y = shortcut @ Wqkv2^T ----
    wave_gemm(sV, 96, 3, p.wqkv, 6, sS, 96, wrow, lane);
    __syncthreads();
    wave_gemm(sS, 96, 3, p.wqkv2, 6, sY, 96, wrow, lane);
    __syncthreads();

    // ---- q = y @ Wq^T -> sU[:,96:192]; Q = elu(q)+1 ----
    wave_gemm(sY, 96, 3, p.wq, 6, sU + 96, 192, wrow, lane);
    __syncthreads();
    for (int i = tid; i < 6144; i += 128) {
        const int r = i / 96, c = i % 96;
        float x = sU[r * 192 + 96 + c];
        sU[r * 192 + 96 + c] = x > 0.f ? x + 1.f : __expf(x);
    }
    __syncthreads();

    // ---- Z = S / (Q . Ksum + eps) ----
    for (int i = tid; i < 512; i += 128) {
        const int r = i >> 3, h = i & 7;
        float acc = 0.f;
#pragma unroll
        for (int d = 0; d < 12; ++d)
            acc += sU[r * 192 + 96 + h * 12 + d] * p.ksum[h * 12 + d];
        sZ[i] = Sval / (acc + 1e-6f);
    }
    __syncthreads();

    // ---- msg_pre = Q @ KVblk -> sU[:,0:96]; scale by Z ----
    wave_gemm(sU + 96, 192, 3, p.kvblk, 6, sU, 192, wrow, lane);
    __syncthreads();
    for (int i = tid; i < 6144; i += 128) {
        const int r = i / 96, c = i % 96;
        sU[r * 192 + c] *= sZ[r * 8 + c / 12];
    }
    __syncthreads();

    // ---- msg = msg_pre @ Wmerge^T -> sU[:,96:192]; LN(sa_ln1) ----
    wave_gemm(sU, 192, 3, p.merge, 6, sU + 96, 192, wrow, lane);
    __syncthreads();
    ln_rows(sU, 192, 96, 96, p.saln1w, p.saln1b, tid, 128, 64);
    __syncthreads();

    // ---- sU = [y | msg] ----
    for (int i = tid; i < 6144; i += 128) {
        const int r = i / 96, c = i % 96;
        sU[r * 192 + c] = sY[i];
    }
    __syncthreads();

    // ---- h1 = [y|msg] @ W1^T (192->192); relu ----
    wave_gemm(sU, 192, 6, p.w1, 12, sV, 192, wrow, lane);
    __syncthreads();
    for (int i = tid; i < 12288; i += 128) sV[i] = fmaxf(sV[i], 0.f);
    __syncthreads();

    // ---- h = h1 @ W2^T (192->96) -> sU[:,96:192]; LN(sa_ln2) ----
    wave_gemm(sV, 192, 6, p.w2, 6, sU + 96, 192, wrow, lane);
    __syncthreads();
    ln_rows(sU, 192, 96, 96, p.saln2w, p.saln2b, tid, 128, 64);
    __syncthreads();

    // ---- x = y + h -> cols 0:96; then cols 96:192 = shortcut; LN2 (192) ----
    for (int i = tid; i < 6144; i += 128) {
        const int r = i / 96, c = i % 96;
        sU[r * 192 + c] = sU[r * 192 + c] + sU[r * 192 + 96 + c];
    }
    __syncthreads();
    for (int i = tid; i < 6144; i += 128) {
        const int r = i / 96, c = i % 96;
        sU[r * 192 + 96 + c] = sS[i];
    }
    __syncthreads();
    ln_rows(sU, 192, 0, 192, p.ln2w, p.ln2b, tid, 128, 64);
    __syncthreads();

    // ---- f = LN @ fc1^T + b1; gelu(exact erf) ----
    wave_gemm(sU, 192, 6, p.fc1, 12, sV, 192, wrow, lane);
    __syncthreads();
    for (int i = tid; i < 12288; i += 128) {
        float x = sV[i] + p.fc1b[i % 192];
        sV[i] = 0.5f * x * (1.f + erff(x * 0.70710678118654752f));
    }
    __syncthreads();

    // ---- out = f @ fc2^T + b2 ----
    wave_gemm(sV, 192, 6, p.fc2, 6, sU, 192, wrow, lane);
    __syncthreads();
    for (int i = tid; i < 6144; i += 128) {
        const int r = i / 96, c = i % 96;
        p.out[(tile0 + r) * 96 + c] = sU[r * 192 + c] + p.fc2b[c];
    }
}

// ---------------------------------------------------------------------------
// Launch
// ---------------------------------------------------------------------------
extern "C" void kernel_launch(void* const* d_in, const int* in_sizes, int n_in,
                              void* d_out, int out_size, void* d_ws, size_t ws_size,
                              hipStream_t stream)
{
    const float* mr   = (const float*)d_in[0];
    const float* warp = (const float*)d_in[2];
    const int L = in_sizes[0] / 96;
    const int S = in_sizes[2] / 96;

    float* ws  = (float*)d_ws;
    u32*   wsu = (u32*)d_ws;

    // ws layout (4-byte units)
    const int OFF_KV = 0, OFF_KSUM = 1152;
    const int off_wqkv  = 1248;
    const int off_wqkv2 = off_wqkv + 4608;
    const int off_wq    = off_wqkv2 + 4608;
    const int off_wk    = off_wq + 4608;
    const int off_wv    = off_wk + 4608;
    const int off_merge = off_wv + 4608;
    const int off_w1    = off_merge + 4608;   // 18432
    const int off_w2    = off_w1 + 18432;     // 9216
    const int off_fc1   = off_w2 + 9216;      // 18432
    const int off_fc2   = off_fc1 + 18432;    // 9216
    const int off_kvblk = off_fc2 + 9216;     // 4608

    // 1) zero KV/Ksum accumulators (graph-replay safe)
    zero_kernel<<<(1248 + 255) / 256, 256, 0, stream>>>(ws, 1248);

    // 2) pack weights into WMMA B-fragment layout (L2-resident thereafter)
    auto pack = [&](const void* W, int off, int Kt, int Nt) {
        const int total = (Kt / 32) * (Nt / 16) * 256;
        pack_weight_kernel<<<(total + 255) / 256, 256, 0, stream>>>(
            (const float*)W, wsu + off, Kt, Nt);
    };
    pack(d_in[3],  off_wqkv,  96, 96);   // w_qkv
    pack(d_in[4],  off_wqkv2, 96, 96);   // w_qkv2
    pack(d_in[13], off_wq,    96, 96);   // sa_wq
    pack(d_in[14], off_wk,    96, 96);   // sa_wk
    pack(d_in[15], off_wv,    96, 96);   // sa_wv
    pack(d_in[16], off_merge, 96, 96);   // sa_merge
    pack(d_in[17], off_w1,    192, 192); // sa_mlp_w1
    pack(d_in[18], off_w2,    192, 96);  // sa_mlp_w2
    pack(d_in[9],  off_fc1,   192, 192); // mlp_fc1_w
    pack(d_in[11], off_fc2,   192, 96);  // mlp_fc2_w

    // 3) linear-attention global statistics
    attn_stats_kernel<<<S / 64, 128, 0, stream>>>(
        warp, wsu + off_wk, wsu + off_wv, ws + OFF_KV, ws + OFF_KSUM,
        1.f / (float)S, (unsigned)S);

    // 4) pack KV into block-diagonal B fragments
    pack_kv_kernel<<<18, 256, 0, stream>>>(ws + OFF_KV, wsu + off_kvblk);

    // 5) fused main pipeline
    P2 p;
    p.mr = mr;
    p.ln1w = (const float*)d_in[5];  p.ln1b = (const float*)d_in[6];
    p.ln2w = (const float*)d_in[7];  p.ln2b = (const float*)d_in[8];
    p.fc1b = (const float*)d_in[10]; p.fc2b = (const float*)d_in[12];
    p.saln1w = (const float*)d_in[19]; p.saln1b = (const float*)d_in[20];
    p.saln2w = (const float*)d_in[21]; p.saln2b = (const float*)d_in[22];
    p.ksum = ws + OFF_KSUM;
    p.wqkv = wsu + off_wqkv;   p.wqkv2 = wsu + off_wqkv2;
    p.wq = wsu + off_wq;       p.merge = wsu + off_merge;
    p.kvblk = wsu + off_kvblk; p.w1 = wsu + off_w1; p.w2 = wsu + off_w2;
    p.fc1 = wsu + off_fc1;     p.fc2 = wsu + off_fc2;
    p.out = (float*)d_out;

    fused_main<<<L / 64, 128, 0, stream>>>(p, (float)S, (unsigned)L);
}
